// VideoEncoder_26233660244652
// MI455X (gfx1250) — matrix operand order repaired and checked by reading, not compiled, not run on hardware
//
#include <hip/hip_runtime.h>

// ---------------- problem constants (from reference) ----------------
#define HEADS   4
#define HD      64          // head dim
#define HID     256         // hidden
#define SEQ_T   2048
#define BATCH   2
#define ROWS    (BATCH * SEQ_T)   // 4096
#define WIN     15
#define NBR     (2 * WIN + 1)     // 31
#define DIN     2048

// ---------------- WMMA GEMM tiling ----------------
#define BM 128
#define BN 64
#define BK 32
// LDS layout: padded 80-byte row stride (64B data + 16B pad):
//  - 16B-aligned so async b128 writes are aligned
//  - bank stride 20 -> 16 consecutive rows hit 16 distinct banks
#define ROWB  80
#define ABUF  (BM * ROWB)          // 10240 B per A buffer
#define BBUF  (BN * ROWB)          // 5120 B per B buffer
#define A0OFF 0
#define A1OFF ABUF
#define B0OFF (2 * ABUF)
#define B1OFF (2 * ABUF + BBUF)
#define LDS_BYTES (2 * ABUF + 2 * BBUF)   // 30720

typedef __attribute__((ext_vector_type(16))) __bf16    v16bf;
typedef __attribute__((ext_vector_type(8)))  float     v8f;
typedef __attribute__((ext_vector_type(8)))  unsigned  v8u;

__device__ __forceinline__ unsigned short f2bf(float a) {
    __bf16 h = (__bf16)a;                       // hardware f32->bf16 cvt
    return __builtin_bit_cast(unsigned short, h);
}

// gfx1250 async copy: global -> LDS, 16 bytes per lane, tracked by ASYNCcnt.
__device__ __forceinline__ void async_copy_b128(unsigned lds_off, unsigned long long gaddr) {
    asm volatile("global_load_async_to_lds_b128 %0, %1, off"
                 :: "v"(lds_off), "v"(gaddr) : "memory");
}
__device__ __forceinline__ void wait_async_le3() {
    asm volatile("s_wait_asynccnt 0x3" ::: "memory");
}
__device__ __forceinline__ void wait_async_0() {
    asm volatile("s_wait_asynccnt 0x0" ::: "memory");
}

// C[M,N] = A[M,K] * Bt[N,K]^T (+ bias[N]); A,Bt bf16 row-major, C f32.
// M % 128 == 0, N % 64 == 0, K % 32 == 0.
__global__ __launch_bounds__(256)
void gemm_bf16_wmma(const unsigned short* __restrict__ A,
                    const unsigned short* __restrict__ Bt,
                    const float* __restrict__ bias, float* __restrict__ C,
                    int M, int N, int K)
{
    __shared__ __attribute__((aligned(128))) unsigned char lds_raw[LDS_BYTES];

    const int tid  = threadIdx.x;
    const int wid  = tid >> 5;
    const int lane = tid & 31;
    const int wm   = wid >> 1;          // 0..3 : wave M tile
    const int wn   = wid & 1;           // 0..1 : wave N tile
    const int bm0  = blockIdx.x * BM;
    const int bn0  = blockIdx.y * BN;

    const int r  = lane & 15;
    const int hi = lane >> 4;

    const unsigned lds_base = (unsigned)(uintptr_t)(void*)lds_raw; // LDS byte offset

    v8f acc[2][2];
    #pragma unroll
    for (int i = 0; i < 2; i++)
        #pragma unroll
        for (int j = 0; j < 2; j++)
            #pragma unroll
            for (int e = 0; e < 8; e++) acc[i][j][e] = 0.0f;

    // Issue one K-tile (A: 128x32 bf16, B: 64x32 bf16) of async b128 copies.
    // Exactly 3 per thread -> per-wave ASYNCcnt bookkeeping is 3 per tile.
    auto issue_tile = [&](int t, int buf) {
        const int k0 = t * BK;
        #pragma unroll
        for (int i = 0; i < 2; i++) {              // A: 512 chunks of 16B
            int idx = tid + i * 256;
            int rr = idx >> 2, cc = idx & 3;
            unsigned lds = lds_base + (buf ? A1OFF : A0OFF) + rr * ROWB + cc * 16;
            unsigned long long g = (unsigned long long)(uintptr_t)
                (A + (size_t)(bm0 + rr) * K + k0 + cc * 8);
            async_copy_b128(lds, g);
        }
        {                                          // B: 256 chunks of 16B
            int rr = tid >> 2, cc = tid & 3;
            unsigned lds = lds_base + (buf ? B1OFF : B0OFF) + rr * ROWB + cc * 16;
            unsigned long long g = (unsigned long long)(uintptr_t)
                (Bt + (size_t)(bn0 + rr) * K + k0 + cc * 8);
            async_copy_b128(lds, g);
        }
    };

    const int nk = K / BK;
    issue_tile(0, 0);
    if (nk > 1) issue_tile(1, 1);

    for (int i = 0; i < nk; ++i) {
        if (i + 1 < nk) wait_async_le3();   // tile i resident, tile i+1 in flight
        else            wait_async_0();     // last tile: drain
        __syncthreads();

        const int buf = i & 1;
        const unsigned char* abase = lds_raw + (buf ? A1OFF : A0OFF);
        const unsigned char* bbase = lds_raw + (buf ? B1OFF : B0OFF);

        // ---- fragments (contiguous k order, same for A and B) ----
        v16bf afrag[2], bfrag[2];
        #pragma unroll
        for (int mi = 0; mi < 2; mi++) {
            const unsigned char* arow = abase + (wm * 32 + mi * 16 + r) * ROWB;
            v8u t;
            #pragma unroll
            for (int v = 0; v < 8; v++) {
                // lanes 0-15: K = 0..15; lanes 16-31: K = 16..31
                int kk = (hi ? 16 : 0) + 2 * v;
                t[v] = *reinterpret_cast<const unsigned*>(arow + 2 * kk);
            }
            afrag[mi] = __builtin_bit_cast(v16bf, t);
        }
        #pragma unroll
        for (int ni = 0; ni < 2; ni++) {
            const unsigned char* brow = bbase + (wn * 32 + ni * 16 + r) * ROWB;
            v8u t;
            #pragma unroll
            for (int v = 0; v < 8; v++) {
                // lanes 0-15: K = 0..15; lanes 16-31: K = 16..31
                int kk = (hi ? 16 : 0) + 2 * v;
                t[v] = *reinterpret_cast<const unsigned*>(brow + 2 * kk);
            }
            bfrag[ni] = __builtin_bit_cast(v16bf, t);
        }

        // ---- 2x2 WMMA per wave per K-step ----
        #pragma unroll
        for (int mi = 0; mi < 2; mi++)
            #pragma unroll
            for (int ni = 0; ni < 2; ni++)
                acc[mi][ni] = __builtin_amdgcn_wmma_f32_16x16x32_bf16(
                    false, afrag[mi], false, bfrag[ni],
                    (short)0, acc[mi][ni], false, false);

        __syncthreads();                    // everyone done reading buf
        if (i + 2 < nk) issue_tile(i + 2, buf);
    }

    // ---- epilogue: D layout lane = N%16, VGPR e -> M = e + 8*hi ----
    #pragma unroll
    for (int mi = 0; mi < 2; mi++) {
        #pragma unroll
        for (int ni = 0; ni < 2; ni++) {
            int col = bn0 + wn * 32 + ni * 16 + r;
            float bv = bias ? bias[col] : 0.0f;
            #pragma unroll
            for (int e = 0; e < 8; e++) {
                int rowg = bm0 + wm * 32 + mi * 16 + hi * 8 + e;
                C[(size_t)rowg * N + col] = acc[mi][ni][e] + bv;
            }
        }
    }
}

// f32 -> bf16 flat convert; n % 1024 == 0, 4 elements per thread.
__global__ __launch_bounds__(256)
void cvt_bf16_kernel(const float* __restrict__ in, unsigned short* __restrict__ out, int n)
{
    int i = (blockIdx.x * 256 + threadIdx.x) * 4;
    if (i >= n) return;
    const float4 v = *reinterpret_cast<const float4*>(in + i);
    ushort4 o;
    o.x = f2bf(v.x); o.y = f2bf(v.y); o.z = f2bf(v.z); o.w = f2bf(v.w);
    *reinterpret_cast<ushort4*>(out + i) = o;
}

// W[K,N] f32 -> Wt[N,K] bf16 (transpose); launched over K*N elements.
__global__ __launch_bounds__(256)
void cvtT_bf16_kernel(const float* __restrict__ W, unsigned short* __restrict__ Wt,
                      int K, int N)
{
    int idx = blockIdx.x * 256 + threadIdx.x;
    if (idx >= K * N) return;
    int k = idx / N, n = idx - k * N;
    Wt[(size_t)n * K + k] = f2bf(W[idx]);
}

// e_src/e_dst: per (row, head) dot of hW head-slice with a_src / a_dst.
__global__ __launch_bounds__(128)
void esd_kernel(const float* __restrict__ hw, const float* __restrict__ asrc,
                const float* __restrict__ adst, float* __restrict__ esrc,
                float* __restrict__ edst)
{
    int row  = blockIdx.x * 4 + (threadIdx.x >> 5);
    int lane = threadIdx.x & 31;
    if (row >= ROWS) return;
    int head = lane >> 3;
    const float* hr = hw + (size_t)row * HID + lane * 8;
    const float* as = asrc + head * HD + (lane & 7) * 8;
    const float* ad = adst + head * HD + (lane & 7) * 8;
    float ss = 0.f, sd = 0.f;
    #pragma unroll
    for (int c = 0; c < 8; c++) { float h = hr[c]; ss += h * as[c]; sd += h * ad[c]; }
    #pragma unroll
    for (int off = 1; off < 8; off <<= 1) {
        ss += __shfl_xor(ss, off, 32);
        sd += __shfl_xor(sd, off, 32);
    }
    if ((lane & 7) == 0) {
        esrc[row * HEADS + head] = ss;
        edst[row * HEADS + head] = sd;
    }
}

// Banded GAT attention: one wave per (row, head); wave32 softmax over 31 neighbors.
__global__ __launch_bounds__(128)
void attn_kernel(const float* __restrict__ esrc, const float* __restrict__ edst,
                 const float* __restrict__ hw, float* __restrict__ out)
{
    int w    = blockIdx.x * 4 + (threadIdx.x >> 5);
    int lane = threadIdx.x & 31;
    int row  = w >> 2;
    int head = w & 3;
    int b    = row / SEQ_T;
    int il   = row - b * SEQ_T;

    int  jl    = il + lane - WIN;
    bool valid = (lane < NBR) && (jl >= 0) && (jl < SEQ_T);
    float e = -3.0e38f;
    if (valid) {
        float t = esrc[row * HEADS + head] + edst[(b * SEQ_T + jl) * HEADS + head];
        e = (t > 0.f) ? t : 0.2f * t;                 // leaky_relu slope 0.2
    }
    float m = e;
    #pragma unroll
    for (int off = 1; off < 32; off <<= 1) m = fmaxf(m, __shfl_xor(m, off, 32));
    float p = valid ? __expf(e - m) : 0.f;
    float s = p;
    #pragma unroll
    for (int off = 1; off < 32; off <<= 1) s += __shfl_xor(s, off, 32);
    float a = p / s;

    float acc0 = 0.f, acc1 = 0.f;
    const float* base = hw + (size_t)(b * SEQ_T) * HID + head * HD;
    #pragma unroll
    for (int j = 0; j < NBR; j++) {
        float aj = __shfl(a, j, 32);
        int   jj = il + j - WIN;
        if (aj > 0.f && jj >= 0 && jj < SEQ_T) {
            const float* hr = base + (size_t)jj * HID;
            acc0 += aj * hr[lane];
            acc1 += aj * hr[lane + 32];
        }
    }
    float* o = out + (size_t)row * HID + head * HD;
    o[lane]      = acc0;
    o[lane + 32] = acc1;
}

// Fused (optional residual) + LayerNorm over a 256-wide row; one wave per row.
__global__ __launch_bounds__(128)
void resln_kernel(const float* __restrict__ x, const float* __restrict__ rsd,
                  const float* __restrict__ g, const float* __restrict__ bta,
                  float* __restrict__ out)
{
    int row  = blockIdx.x * 4 + (threadIdx.x >> 5);
    int lane = threadIdx.x & 31;
    const float* xr = x + (size_t)row * HID + lane * 8;
    float v[8];
    #pragma unroll
    for (int c = 0; c < 8; c++) v[c] = xr[c];
    if (rsd) {
        const float* rr = rsd + (size_t)row * HID + lane * 8;
        #pragma unroll
        for (int c = 0; c < 8; c++) v[c] += rr[c];
    }
    float s = 0.f, q = 0.f;
    #pragma unroll
    for (int c = 0; c < 8; c++) { s += v[c]; q += v[c] * v[c]; }
    #pragma unroll
    for (int off = 1; off < 32; off <<= 1) {
        s += __shfl_xor(s, off, 32);
        q += __shfl_xor(q, off, 32);
    }
    float mean = s * (1.0f / HID);
    float var  = q * (1.0f / HID) - mean * mean;
    float rstd = rsqrtf(var + 1e-5f);
    float* o = out + (size_t)row * HID + lane * 8;
    #pragma unroll
    for (int c = 0; c < 8; c++) {
        int ch = lane * 8 + c;
        o[c] = (v[c] - mean) * rstd * g[ch] + bta[ch];
    }
}

extern "C" void kernel_launch(void* const* d_in, const int* in_sizes, int n_in,
                              void* d_out, int out_size, void* d_ws, size_t ws_size,
                              hipStream_t stream)
{
    const float* x   = (const float*)d_in[0];
    // d_in[1] = adj_mask (banded window, implied by WIN): not needed
    const float* Wp  = (const float*)d_in[2];
    const float* bp  = (const float*)d_in[3];
    const float* W0  = (const float*)d_in[4];
    const float* as0 = (const float*)d_in[5];
    const float* ad0 = (const float*)d_in[6];
    const float* g0  = (const float*)d_in[7];
    const float* b0  = (const float*)d_in[8];
    const float* W1  = (const float*)d_in[9];
    const float* as1 = (const float*)d_in[10];
    const float* ad1 = (const float*)d_in[11];
    const float* g1  = (const float*)d_in[12];
    const float* b1  = (const float*)d_in[13];
    const float* gF  = (const float*)d_in[14];
    const float* bF  = (const float*)d_in[15];
    float* out = (float*)d_out;

    // ---- workspace layout ----
    float* ws  = (float*)d_ws;
    float* h   = ws;                                  // ROWS*HID f32
    float* hw  = h   + (size_t)ROWS * HID;
    float* att = hw  + (size_t)ROWS * HID;
    float* es  = att + (size_t)ROWS * HID;
    float* ed  = es  + (size_t)ROWS * HEADS;
    unsigned short* xbf  = (unsigned short*)(ed + (size_t)ROWS * HEADS); // ROWS*DIN
    unsigned short* hbf  = xbf + (size_t)ROWS * DIN;                     // ROWS*HID
    unsigned short* WpT  = hbf + (size_t)ROWS * HID;                     // HID*DIN
    unsigned short* W0T  = WpT + (size_t)HID * DIN;                      // HID*HID
    unsigned short* W1T  = W0T + (size_t)HID * HID;

    dim3 gGemm(ROWS / BM, HID / BN);    // 32 x 4
    dim3 bGemm(256);

    // ---- preconvert to bf16 (x flat; weights transposed to [N][K]) ----
    cvt_bf16_kernel <<<(ROWS * DIN) / 1024, 256, 0, stream>>>(x, xbf, ROWS * DIN);
    cvtT_bf16_kernel<<<(DIN * HID) / 256,   256, 0, stream>>>(Wp, WpT, DIN, HID);
    cvtT_bf16_kernel<<<(HID * HID) / 256,   256, 0, stream>>>(W0, W0T, HID, HID);
    cvtT_bf16_kernel<<<(HID * HID) / 256,   256, 0, stream>>>(W1, W1T, HID, HID);

    // h = x @ Wp + bp      (M=4096, N=256, K=2048)
    gemm_bf16_wmma<<<gGemm, bGemm, 0, stream>>>(xbf, WpT, bp, h, ROWS, HID, DIN);

    // ---- GAT block 0 ----
    cvt_bf16_kernel<<<(ROWS * HID) / 1024, 256, 0, stream>>>(h, hbf, ROWS * HID);
    gemm_bf16_wmma<<<gGemm, bGemm, 0, stream>>>(hbf, W0T, nullptr, hw, ROWS, HID, HID);
    esd_kernel  <<<ROWS / 4,         128, 0, stream>>>(hw, as0, ad0, es, ed);
    attn_kernel <<<ROWS * HEADS / 4, 128, 0, stream>>>(es, ed, hw, att);
    resln_kernel<<<ROWS / 4,         128, 0, stream>>>(h, att, g0, b0, h);

    // ---- GAT block 1 ----
    cvt_bf16_kernel<<<(ROWS * HID) / 1024, 256, 0, stream>>>(h, hbf, ROWS * HID);
    gemm_bf16_wmma<<<gGemm, bGemm, 0, stream>>>(hbf, W1T, nullptr, hw, ROWS, HID, HID);
    esd_kernel  <<<ROWS / 4,         128, 0, stream>>>(hw, as1, ad1, es, ed);
    attn_kernel <<<ROWS * HEADS / 4, 128, 0, stream>>>(es, ed, hw, att);
    resln_kernel<<<ROWS / 4,         128, 0, stream>>>(h, att, g1, b1, h);

    // ---- final LayerNorm ----
    resln_kernel<<<ROWS / 4, 128, 0, stream>>>(h, nullptr, gF, bF, out);
}
